// PatchWiseConvBlock3D_72834055405654
// MI455X (gfx1250) — matrix operand order repaired and checked by reading, not compiled, not run on hardware
//
#include <hip/hip_runtime.h>
#include <math.h>

// CDNA5 / gfx1250: wave32, WMMA 16x16x32 f16 -> f32
typedef __attribute__((ext_vector_type(16))) _Float16 v16h;
typedef __attribute__((ext_vector_type(8)))  float    v8f;

#define NPD    31
#define NP     (31*31*31)      // 29791 patches
#define WPB    4               // waves (=patches) per 128-thread block
#define KTOT   432             // 16 ch * 27 taps
#define KPAD   448             // pad K to multiple of 32
#define KSTEPS 14              // 448/32
#define PPADN  3600            // 16*6*6*6 = 3456 padded patch halves + 144 zero tail
#define EPSBN  1e-5f

// ---------------------------------------------------------------- zero output
__global__ __launch_bounds__(256)
void k_zero(float* __restrict__ p, int n) {
    int i = blockIdx.x * blockDim.x + threadIdx.x;
    if (i < n) p[i] = 0.f;
}

// -------------------------------------------------- fused per-patch pipeline
// conv3x3x3(pad1, per-patch zero pad) + BN + ReLU + SE + avgpool2 + fold-accum
__global__ __launch_bounds__(128)
void k_patch(const float* __restrict__ x,
             const float* __restrict__ conv_w,
             const float* __restrict__ bn_g, const float* __restrict__ bn_b,
             const float* __restrict__ bn_m, const float* __restrict__ bn_v,
             const float* __restrict__ se_w1, const float* __restrict__ se_w2,
             float* __restrict__ acc_out)
{
    __shared__ _Float16      sW[32 * KPAD];      // weights f16, row co, K-contig
    __shared__ unsigned short sLUT[KPAD];         // K -> padded-patch half offset
    __shared__ float          sBN[64];            // scale[32], shift[32]
    __shared__ float          sSE1[4 * 32];
    __shared__ float          sSE2[32 * 4];
    __shared__ _Float16       sPatch[WPB][PPADN]; // padded patch per wave + zero tail
    __shared__ float          sDesc[WPB][32];

    const int tid = threadIdx.x;

    // ---- cooperative staging (whole block) ----
    for (int idx = tid; idx < 32 * KPAD; idx += 128) {
        int m = idx / KPAD, k = idx % KPAD;
        float w = 0.f;
        if (k < KTOT) { int ci = k / 27, r = k % 27; w = conv_w[(m * 16 + ci) * 27 + r]; }
        sW[idx] = (_Float16)w;
    }
    for (int k = tid; k < KPAD; k += 128) {
        unsigned short off = 3456;                 // zero-tail region
        if (k < KTOT) {
            int ci = k / 27, r = k % 27;
            int kd = r / 9, kh = (r % 9) / 3, kw = r % 3;
            off = (unsigned short)(ci * 216 + kd * 36 + kh * 6 + kw);
        }
        sLUT[k] = off;
    }
    if (tid < 32) {
        float inv = bn_g[tid] * rsqrtf(bn_v[tid] + EPSBN);
        sBN[tid]      = inv;
        sBN[32 + tid] = bn_b[tid] - bn_m[tid] * inv;
    }
    if (tid < 128) { sSE1[tid] = se_w1[tid]; sSE2[tid] = se_w2[tid]; }
    __syncthreads();

    const int wave = tid >> 5, lane = tid & 31;
    const int pid  = blockIdx.x * WPB + wave;      // one patch per wave
    if (pid >= NP) return;

    const int pz = pid / (NPD * NPD);
    const int prem = pid % (NPD * NPD);
    const int py = prem / NPD, px = prem % NPD;
    const int d0 = pz * 2, h0 = py * 2, w0 = px * 2;

    // ---- stage this wave's patch into LDS as f16, with 1-voxel zero shell ----
    _Float16* pp = &sPatch[wave][0];
    for (int i = lane; i < PPADN; i += 32) pp[i] = (_Float16)0.f;
    for (int e = lane; e < 1024; e += 32) {        // 16ch * 4^3
        int ci = e >> 6, s = e & 63;
        int zd = s >> 4, zy = (s >> 2) & 3, zx = s & 3;
        float v = x[(((ci * 64) + (d0 + zd)) * 64 + (h0 + zy)) * 64 + (w0 + zx)];
        pp[ci * 216 + (zd + 1) * 36 + (zy + 1) * 6 + (zx + 1)] = (_Float16)v;
    }
    // DS executes in order within a wave, but fence the compiler + outstanding DS
    asm volatile("s_wait_dscnt 0" ::: "memory");

    // ---- per-lane fragment geometry (ISA 7.12.2 layouts) ----
    const int  m16    = lane & 15;
    const bool hihalf = lane >= 16;
    const int  aklo   = hihalf ? 8 : 0;            // A: K sub-block per half-wave
    const int  bko    = hihalf ? 16 : 0;           // unused; B indexes K with aklo, same as A
    const int  arow0  = m16 * KPAD;                // A rows: co = m16 (M-tile 0)
    const int  arow1  = (16 + m16) * KPAD;         //         co = 16+m16 (M-tile 1)
    const int  sp_base = (m16 >> 2) * 6 + (m16 & 3); // oh*6 + ow for column n

    v8f zero8 = {0.f, 0.f, 0.f, 0.f, 0.f, 0.f, 0.f, 0.f};
    v8f acc[2][4];
    #pragma unroll
    for (int mt = 0; mt < 2; ++mt)
        #pragma unroll
        for (int nt = 0; nt < 4; ++nt) acc[mt][nt] = zero8;

    // ---- GEMM: D(32co x 64sp) = W(32 x 432) * im2col(432 x 64) ----
    for (int kk = 0; kk < KSTEPS; ++kk) {
        const int k0 = kk * 32;
        v16h a0, a1;
        #pragma unroll
        for (int h = 0; h < 8; ++h) {              // halves 0..7 = K+aklo, 8..15 = K+16+aklo
            a0[h]     = sW[arow0 + k0 + aklo + h];
            a0[8 + h] = sW[arow0 + k0 + 16 + aklo + h];
            a1[h]     = sW[arow1 + k0 + aklo + h];
            a1[8 + h] = sW[arow1 + k0 + 16 + aklo + h];
        }
        v16h b0, b1, b2, b3;                       // N-tiles = output depth od = 0..3
        #pragma unroll
        for (int h = 0; h < 16; ++h) {
            int off = (int)sLUT[k0 + aklo + h + (h >= 8 ? 8 : 0)] + sp_base;
            b0[h] = pp[off];
            b1[h] = pp[off + 36];
            b2[h] = pp[off + 72];
            b3[h] = pp[off + 108];
        }
        acc[0][0] = __builtin_amdgcn_wmma_f32_16x16x32_f16(false, a0, false, b0, (short)0, acc[0][0], false, false);
        acc[0][1] = __builtin_amdgcn_wmma_f32_16x16x32_f16(false, a0, false, b1, (short)0, acc[0][1], false, false);
        acc[0][2] = __builtin_amdgcn_wmma_f32_16x16x32_f16(false, a0, false, b2, (short)0, acc[0][2], false, false);
        acc[0][3] = __builtin_amdgcn_wmma_f32_16x16x32_f16(false, a0, false, b3, (short)0, acc[0][3], false, false);
        acc[1][0] = __builtin_amdgcn_wmma_f32_16x16x32_f16(false, a1, false, b0, (short)0, acc[1][0], false, false);
        acc[1][1] = __builtin_amdgcn_wmma_f32_16x16x32_f16(false, a1, false, b1, (short)0, acc[1][1], false, false);
        acc[1][2] = __builtin_amdgcn_wmma_f32_16x16x32_f16(false, a1, false, b2, (short)0, acc[1][2], false, false);
        acc[1][3] = __builtin_amdgcn_wmma_f32_16x16x32_f16(false, a1, false, b3, (short)0, acc[1][3], false, false);
    }

    // ---- BN + ReLU + spatial-mean descriptor (half-wave = one co group) ----
    #pragma unroll
    for (int mt = 0; mt < 2; ++mt) {
        #pragma unroll
        for (int r = 0; r < 8; ++r) {
            int co = mt * 16 + r + (hihalf ? 8 : 0);   // D layout: M = r (+8 hi half)
            float sc = sBN[co], sh = sBN[32 + co];
            float s = 0.f;
            #pragma unroll
            for (int nt = 0; nt < 4; ++nt) {
                float y = fmaxf(acc[mt][nt][r] * sc + sh, 0.f);
                acc[mt][nt][r] = y;
                s += y;
            }
            s += __shfl_xor(s, 1, 32); s += __shfl_xor(s, 2, 32);
            s += __shfl_xor(s, 4, 32); s += __shfl_xor(s, 8, 32);
            if (m16 == 0) sDesc[wave][co] = s * (1.f / 64.f);
        }
    }
    asm volatile("s_wait_dscnt 0" ::: "memory");

    // ---- SE MLP (tiny, each lane computes what it needs) ----
    float hid[4];
    #pragma unroll
    for (int hh = 0; hh < 4; ++hh) {
        float t = 0.f;
        for (int c = 0; c < 32; ++c) t += sDesc[wave][c] * sSE1[hh * 32 + c];
        hid[hh] = fmaxf(t, 0.f);
    }
    float attnv[2][8];
    #pragma unroll
    for (int mt = 0; mt < 2; ++mt)
        #pragma unroll
        for (int r = 0; r < 8; ++r) {
            int co = mt * 16 + r + (hihalf ? 8 : 0);
            float t = hid[0] * sSE2[co * 4 + 0] + hid[1] * sSE2[co * 4 + 1]
                    + hid[2] * sSE2[co * 4 + 2] + hid[3] * sSE2[co * 4 + 3];
            attnv[mt][r] = 1.f / (1.f + __expf(-t));
        }

    // ---- 2x2x2 avg-pool + fold accumulation (stride-1 overlap handled by atomics) ----
    const int  ph  = (m16 >> 3) & 1;               // oh/2
    const int  pw  = (m16 >> 1) & 1;               // ow/2
    const bool rep = ((m16 & 5) == 0);             // ow even && oh even
    #pragma unroll
    for (int mt = 0; mt < 2; ++mt)
        #pragma unroll
        for (int r = 0; r < 8; ++r) {
            int co = mt * 16 + r + (hihalf ? 8 : 0);
            #pragma unroll
            for (int pd = 0; pd < 2; ++pd) {       // N-tile pairs = depth pairs
                float v = (acc[mt][2 * pd][r] + acc[mt][2 * pd + 1][r]) * attnv[mt][r];
                v += __shfl_xor(v, 1, 32);         // sum over ow pair
                v += __shfl_xor(v, 4, 32);         // sum over oh pair
                if (rep) {
                    int oz = pz + pd, oy = py + ph, ox = px + pw;
                    atomicAdd(&acc_out[((co * 32 + oz) * 32 + oy) * 32 + ox], v * 0.125f);
                }
            }
        }
}

// ------------------------------- fold-normalize + strided 1x1x1 residual + BN
__global__ __launch_bounds__(256)
void k_combine(const float* __restrict__ x,
               const float* __restrict__ down_w,
               const float* __restrict__ dg, const float* __restrict__ db,
               const float* __restrict__ dm, const float* __restrict__ dv,
               float* __restrict__ out)
{
    int i = blockIdx.x * blockDim.x + threadIdx.x;
    if (i >= 32 * 32 * 32) return;
    int z = i >> 10, y = (i >> 5) & 31, xc = i & 31;

    float xin[16];
    #pragma unroll
    for (int ci = 0; ci < 16; ++ci)
        xin[ci] = x[((ci * 64 + 2 * z) * 64 + 2 * y) * 64 + 2 * xc];

    float cz = (z == 0 || z == 31) ? 1.f : 2.f;
    float cy = (y == 0 || y == 31) ? 1.f : 2.f;
    float cx = (xc == 0 || xc == 31) ? 1.f : 2.f;
    float invwt = 1.f / (cz * cy * cx);

    for (int co = 0; co < 32; ++co) {
        float rsum = 0.f;
        #pragma unroll
        for (int ci = 0; ci < 16; ++ci) rsum += xin[ci] * down_w[co * 16 + ci];
        float inv = dg[co] * rsqrtf(dv[co] + EPSBN);
        float res = rsum * inv + (db[co] - dm[co] * inv);
        int oidx = ((co * 32 + z) * 32 + y) * 32 + xc;
        out[oidx] = out[oidx] * invwt + res;
    }
}

extern "C" void kernel_launch(void* const* d_in, const int* in_sizes, int n_in,
                              void* d_out, int out_size, void* d_ws, size_t ws_size,
                              hipStream_t stream) {
    const float* x      = (const float*)d_in[0];
    const float* conv_w = (const float*)d_in[1];
    const float* bn_g   = (const float*)d_in[2];
    const float* bn_b   = (const float*)d_in[3];
    const float* bn_m   = (const float*)d_in[4];
    const float* bn_v   = (const float*)d_in[5];
    const float* se_w1  = (const float*)d_in[6];
    const float* se_w2  = (const float*)d_in[7];
    const float* down_w = (const float*)d_in[8];
    const float* dbn_g  = (const float*)d_in[9];
    const float* dbn_b  = (const float*)d_in[10];
    const float* dbn_m  = (const float*)d_in[11];
    const float* dbn_v  = (const float*)d_in[12];
    float* out = (float*)d_out;

    const int n_out = 32 * 32 * 32 * 32;
    k_zero<<<(n_out + 255) / 256, 256, 0, stream>>>(out, n_out);
    k_patch<<<(NP + WPB - 1) / WPB, 128, 0, stream>>>(x, conv_w, bn_g, bn_b, bn_m, bn_v,
                                                      se_w1, se_w2, out);
    k_combine<<<(32 * 32 * 32 + 255) / 256, 256, 0, stream>>>(x, down_w, dbn_g, dbn_b,
                                                              dbn_m, dbn_v, out);
}
